// ST_BLOCK_2_25245817766080
// MI455X (gfx1250) — matrix operand order repaired and checked by reading, not compiled, not run on hardware
//
#include <hip/hip_runtime.h>
#include <cstdint>

// ---------------------------------------------------------------------------
// Types for CDNA5 WMMA
// ---------------------------------------------------------------------------
typedef __attribute__((ext_vector_type(16))) _Float16 v16h;
typedef __attribute__((ext_vector_type(8)))  float    v8f;

__device__ __forceinline__ float sigf(float x)  { return 1.0f / (1.0f + __expf(-x)); }
__device__ __forceinline__ float leakyf(float x){ return x >= 0.0f ? x : 0.01f * x; }

static const long NELT  = 15728640L;     // B*CO*N*L
static const long ADJSZ = 2097152L;      // B*N*N
static const float EPSV = 1e-5f;

// 16-byte async global->LDS copy (CDNA5 GLOBAL_LOAD_ASYNC_TO_LDS_B128).
// Per-lane: LDS[lds_off .. +15] = MEM[gptr .. +15]; tracked by ASYNCcnt.
__device__ __forceinline__ void async_cp16(unsigned lds_off, const _Float16* gptr)
{
    asm volatile("global_load_async_to_lds_b128 %0, %1, off"
                 :: "v"(lds_off), "v"(gptr) : "memory");
}
__device__ __forceinline__ void async_wait0()
{
    asm volatile("s_wait_asynccnt 0x0" ::: "memory");
}
__device__ __forceinline__ unsigned lds_off32(const void* p)
{
    return (unsigned)(unsigned long long)p;   // low 32 bits of generic ptr = LDS offset
}

// ---------------------------------------------------------------------------
// WMMA GEMM:  C[M,N] = A[M,K] * B[K,N]   (f16 operands, f32 accumulate)
// A: f16 row-major with grouped-affine rows: elem(r,k) = A[r*lda + (r/grp)*skip + k]
// B: f16 K-MAJOR panel: elem(k,n) = Bt[n*ldbt + k]
// Block: 128 threads (4 waves), tile 64(M) x 128(N), K staged 32 at a time,
// double-buffered LDS, staging via async global->LDS b128 copies.
// All 8 B fragments are loaded into registers before the WMMA burst so the
// eight v_wmma issue back-to-back behind a single DS wait.
//   mode 0: ((float*)C)[r*ldc + c] = acc
//   mode 1: Chebyshev scatter (f16 out): c -> (cc=c/60, l=c%60),
//           ((f16*)C)[((p0 + r)*62 + l + 1)*192 + cc*3 + p1] = acc
//   mode 2: ((f16*)C)[r*ldc + c] = 2*acc - (r==c)
// Requirements: M%64==0, N%128==0, K%32==0; 16B alignment of all panels.
// ---------------------------------------------------------------------------
__global__ __launch_bounds__(128)
void gemm_wmma(const _Float16* __restrict__ A, const _Float16* __restrict__ Bt,
               void* __restrict__ C,
               int K, int lda, long grp, long skip, int ldbt, int ldc,
               int mode, long p0, long p1)
{
    __shared__ _Float16 As[2][64 * 32];    // [row][k]
    __shared__ _Float16 Bs[2][128 * 32];   // [col][k]

    const int tid  = threadIdx.x;
    const int lane = tid & 31;
    const int wv   = tid >> 5;
    const int hsel = lane >> 4;
    const int mrow = lane & 15;
    const long rowBase = (long)blockIdx.x * 64;
    const long colBase = (long)blockIdx.y * 128;

    // Per-thread staging addresses (element units), hoisted out of the k-loop.
    const int akc  = tid & 3;             // 8-f16 (16B) chunk within a 32-k row
    const int srow = tid >> 2;            // 0..31
    long abase[2];
#pragma unroll
    for (int e = 0; e < 2; ++e) {
        long gr = rowBase + e * 32 + srow;
        abase[e] = gr * (long)lda + (gr / grp) * skip + akc * 8;
    }
    long bbase[4];
#pragma unroll
    for (int e = 0; e < 4; ++e)
        bbase[e] = (colBase + e * 32 + srow) * (long)ldbt + akc * 8;

    auto stage = [&](int buf, int k0) {
#pragma unroll
        for (int e = 0; e < 2; ++e)
            async_cp16(lds_off32(&As[buf][(e * 32 + srow) * 32 + akc * 8]),
                       A + abase[e] + k0);
#pragma unroll
        for (int e = 0; e < 4; ++e)
            async_cp16(lds_off32(&Bs[buf][(e * 32 + srow) * 32 + akc * 8]),
                       Bt + bbase[e] + k0);
    };

    v8f zero = {0.f, 0.f, 0.f, 0.f, 0.f, 0.f, 0.f, 0.f};
    v8f acc[8];
#pragma unroll
    for (int i = 0; i < 8; ++i) acc[i] = zero;

    const int nkt = K >> 5;
    stage(0, 0);
    async_wait0();
    __syncthreads();

    for (int kt = 0; kt < nkt; ++kt) {
        int buf = kt & 1;
        bool more = (kt + 1 < nkt);
        if (more) stage(buf ^ 1, (kt + 1) * 32);   // async copies overlap compute

        // Load A fragment and ALL 8 B fragments first, then burst the WMMAs.
        union { v16h v; _Float16 h[16]; } af, bf[8];
        const int arow = wv * 16 + mrow;
#pragma unroll
        for (int i = 0; i < 8; ++i) {
            int kb = 2 * i + 16 * hsel;
            af.h[2 * i]     = As[buf][arow * 32 + kb];
            af.h[2 * i + 1] = As[buf][arow * 32 + kb + 1];
        }
#pragma unroll
        for (int nt = 0; nt < 8; ++nt) {
            int bcol = nt * 16 + mrow;
#pragma unroll
            for (int i = 0; i < 8; ++i) {
                int kb = 2 * i + 16 * hsel;
                bf[nt].h[2 * i]     = Bs[buf][bcol * 32 + kb];
                bf[nt].h[2 * i + 1] = Bs[buf][bcol * 32 + kb + 1];
            }
        }
#pragma unroll
        for (int nt = 0; nt < 8; ++nt) {
            acc[nt] = __builtin_amdgcn_wmma_f32_16x16x32_f16(
                false, af.v, false, bf[nt].v, (short)0, acc[nt], false, false);
        }
        if (more) async_wait0();
        __syncthreads();
    }

    // ---- epilogue ----
#pragma unroll
    for (int nt = 0; nt < 8; ++nt) {
        long cg = colBase + nt * 16 + mrow;
        long cc = 0, l = 0;
        if (mode == 1) { cc = cg / 60; l = cg % 60; }
#pragma unroll
        for (int r = 0; r < 8; ++r) {
            long rg = rowBase + wv * 16 + r + 8 * hsel;
            float v = acc[nt][r];
            if (mode == 0) {
                ((float*)C)[rg * (long)ldc + cg] = v;
            } else if (mode == 1) {
                ((_Float16*)C)[((p0 + rg) * 62 + l + 1) * 192 + cc * 3 + p1] = (_Float16)v;
            } else {
                ((_Float16*)C)[rg * (long)ldc + cg] =
                    (_Float16)(2.0f * v - ((rg == cg) ? 1.0f : 0.0f));
            }
        }
    }
}

// ---------------------------------------------------------------------------
// Pointwise / small kernels
// ---------------------------------------------------------------------------
__global__ void zero_kernel(float* p, long n)
{
    long i = (long)blockIdx.x * blockDim.x + threadIdx.x;
    if (i < n) p[i] = 0.0f;
}

// f32 -> f16 elementwise
__global__ void cvt_kernel(const float* __restrict__ in, _Float16* __restrict__ out, long n)
{
    long i = (long)blockIdx.x * blockDim.x + threadIdx.x;
    if (i < n) out[i] = (_Float16)in[i];
}

// x_in = conv1x1(x) + b        (B,CO,N,L)
__global__ void conv1_kernel(const float* __restrict__ x, const float* __restrict__ w,
                             const float* __restrict__ bias, float* __restrict__ out)
{
    long i = (long)blockIdx.x * blockDim.x + threadIdx.x;
    if (i >= NELT) return;
    int l = i % 60; long t = i / 60;
    int n = t % 512; t /= 512;
    int o = t % 64;  int b = t / 64;
    float s = bias[o];
    for (int c = 0; c < 16; ++c)
        s += w[o * 16 + c] * x[(((long)b * 16 + c) * 512 + n) * 60 + l];
    out[i] = s;
}

// x1 = leaky_relu(conv1x3(x) + b), pad 1 along L
__global__ void tconv_kernel(const float* __restrict__ x, const float* __restrict__ w,
                             const float* __restrict__ bias, float* __restrict__ out)
{
    long i = (long)blockIdx.x * blockDim.x + threadIdx.x;
    if (i >= NELT) return;
    int l = i % 60; long t = i / 60;
    int n = t % 512; t /= 512;
    int o = t % 64;  int b = t / 64;
    float s = bias[o];
    for (int c = 0; c < 16; ++c) {
        long xb = (((long)b * 16 + c) * 512 + n) * 60;
        for (int d = 0; d < 3; ++d) {
            int ll = l + d - 1;
            if (ll >= 0 && ll < 60)
                s += w[((long)o * 16 + c) * 3 + d] * x[xb + ll];
        }
    }
    out[i] = leakyf(s);
}

// F[b][g16][m][n][c16]  (g<4: S1 path via s3 weights; g>=4: S2 path via s2 weights)
__global__ void satt_f_kernel(const float* __restrict__ x1, const float* __restrict__ w3,
                              const float* __restrict__ w2, float* __restrict__ F)
{
    long i = (long)blockIdx.x * blockDim.x + threadIdx.x;
    if (i >= 4194304L) return;
    int c16 = i % 16; long t = i / 16;
    int n = t % 512; t /= 512;
    int m = t % 4;   t /= 4;
    int g = t % 16;  int b = t / 16;
    int o = c16 * 4 + m;
    float s = 0.0f;
    if (g < 4) {
        for (int c = 0; c < 64; ++c) {
            long xb = (((long)b * 64 + c) * 512 + n) * 60;
            for (int h = 0; h < 12; ++h)
                s += w3[(long)o * 768 + c * 12 + h] * x1[xb + 4 * h + g];
        }
    } else {
        int tt = g - 4;
        for (int c = 0; c < 64; ++c)
            s += w2[o * 64 + c] * x1[(((long)b * 64 + c) * 512 + n) * 60 + 48 + tt];
    }
    F[i] = s;
}

// S[b][g][n][q] = mean_m sigmoid( dot_c16 F1 . F2 )   (f16 out: LSTM GEMM operand)
__global__ void satt_s_kernel(const float* __restrict__ F1, const float* __restrict__ F2,
                              _Float16* __restrict__ S)
{
    long i = (long)blockIdx.x * blockDim.x + threadIdx.x;
    if (i >= 16777216L) return;
    int q = i % 512; long t = i / 512;
    int n = t % 512; t /= 512;
    int g = t % 16;  int b = t / 16;
    float acc = 0.0f;
    for (int m = 0; m < 4; ++m) {
        const float* a  = F1 + (((((long)b * 16 + g) * 4 + m) * 512) + n) * 16;
        const float* bb = F2 + (((((long)b * 16 + g) * 4 + m) * 512) + q) * 16;
        float d = 0.0f;
        for (int c = 0; c < 16; ++c) d += a[c] * bb[c];
        acc += sigf(d);
    }
    S[i] = (_Float16)(acc * 0.25f);
}

// x1tt[b][c*60+l][n] = x1[b][c][n][l]    (f16 K-major B panel for Chebyshev GEMMs)
__global__ void x1tt_kernel(const float* __restrict__ x1, _Float16* __restrict__ x1tt)
{
    long i = (long)blockIdx.x * blockDim.x + threadIdx.x;
    if (i >= NELT) return;
    int l = i % 60; long t = i / 60;
    int n = t % 512; t /= 512;
    int c = t % 64;  int b = t / 64;
    x1tt[((long)b * 3840 + c * 60 + l) * 512 + n] = (_Float16)x1[i];
}

// gcnBt[o][dl*192+ci] = gcn_w[o][ci][0][dl]   (f16 K-major, ldbt=576)
__global__ void gcnw_kernel(const float* __restrict__ w, _Float16* __restrict__ Bg)
{
    long i = (long)blockIdx.x * blockDim.x + threadIdx.x;
    if (i >= 73728L) return;
    int kk = i % 576; long o = i / 576;
    int ci = kk % 192; int dl = kk / 192;
    Bg[i] = (_Float16)w[(o * 192 + ci) * 3 + dl];
}

// LSTM gate pointwise (also produces f16 copy of h for the next recurrent GEMM)
__global__ void lstm_point_kernel(const float* __restrict__ Gx, const float* __restrict__ Gh,
                                  const float* __restrict__ bih, const float* __restrict__ bhh,
                                  float* __restrict__ h, float* __restrict__ c,
                                  _Float16* __restrict__ hH)
{
    long i = (long)blockIdx.x * blockDim.x + threadIdx.x;
    if (i >= 2097152L) return;
    long bn = i / 512, j = i % 512;
    long base = bn * 2048;
    float gi = Gx[base + j]        + Gh[base + j]        + bih[j]        + bhh[j];
    float gf = Gx[base + 512 + j]  + Gh[base + 512 + j]  + bih[512 + j]  + bhh[512 + j];
    float gg = Gx[base + 1024 + j] + Gh[base + 1024 + j] + bih[1024 + j] + bhh[1024 + j];
    float go = Gx[base + 1536 + j] + Gh[base + 1536 + j] + bih[1536 + j] + bhh[1536 + j];
    float cn = sigf(gf) * c[i] + sigf(gi) * tanhf(gg);
    float hn = sigf(go) * tanhf(cn);
    c[i] = cn;
    h[i] = hn;
    hH[i] = (_Float16)hn;
}

// adj_out (f32 to d_out), adj1H = h*supports (f16, GEMM A), adj1HT = transpose (GEMM B)
__global__ void adjfinish_kernel(const float* __restrict__ h, const float* __restrict__ sup,
                                 float* __restrict__ adjout,
                                 _Float16* __restrict__ adj1H, _Float16* __restrict__ adj1HT)
{
    long i = (long)blockIdx.x * blockDim.x + threadIdx.x;
    if (i >= ADJSZ) return;
    int q = i % 512; long t = i / 512;
    int n = t % 512; long b = t / 512;
    float v = h[i];
    adjout[i] = v;
    float a = v * sup[i];
    adj1H[i] = (_Float16)a;
    adj1HT[(b * 512 + q) * 512 + n] = (_Float16)a;
}

// k=0 identity term of Chebyshev into padded GCN layout (f16)
__global__ void k0copy_kernel(const float* __restrict__ x1, _Float16* __restrict__ xgpad)
{
    long i = (long)blockIdx.x * blockDim.x + threadIdx.x;
    if (i >= NELT) return;
    int l = i % 60; long t = i / 60;
    int n = t % 512; t /= 512;
    int c = t % 64;  int b = t / 64;
    xgpad[(((long)b * 512 + n) * 62 + l + 1) * 192 + c * 3] = (_Float16)x1[i];
}

// GLU-ish epilogue: x1b = sigmoid(gate+b) * leaky(filt+b)
__global__ void gcn_epi_kernel(const float* __restrict__ gout, const float* __restrict__ gb,
                               float* __restrict__ x1b)
{
    long i = (long)blockIdx.x * blockDim.x + threadIdx.x;
    if (i >= NELT) return;
    int l = i % 60; long t = i / 60;
    int n = t % 512; t /= 512;
    int c = t % 64;  int b = t / 64;
    long r = ((long)b * 512 + n) * 60 + l;
    float f = gout[r * 128 + c]      + gb[c];
    float g = gout[r * 128 + 64 + c] + gb[64 + c];
    x1b[i] = sigf(g) * leakyf(f);
}

// f1v[b][l][n] = sum_c t_w1[c] * x1b[b,c,n,l]
__global__ void f1v_kernel(const float* __restrict__ x1b, const float* __restrict__ tw1,
                           float* __restrict__ f1v)
{
    long i = (long)blockIdx.x * blockDim.x + threadIdx.x;
    if (i >= 245760L) return;
    int n = i % 512; long t = i / 512;
    int l = t % 60; int b = t / 60;
    float s = 0.0f;
    for (int c = 0; c < 64; ++c)
        s += tw1[c] * x1b[(((long)b * 64 + c) * 512 + n) * 60 + l];
    f1v[i] = s;
}

// f2v[b][c][l] = sum_n t_w2[n] * x1b[b,c,n,l]
__global__ void f2v_kernel(const float* __restrict__ x1b, const float* __restrict__ tw2,
                           float* __restrict__ f2v)
{
    long i = (long)blockIdx.x * blockDim.x + threadIdx.x;
    if (i >= 30720L) return;
    int l = i % 60; long t = i / 60;
    int c = t % 64; int b = t / 64;
    float s = 0.0f;
    long xb = ((long)b * 64 + c) * 512;
    for (int n = 0; n < 512; ++n)
        s += tw2[n] * x1b[(xb + n) * 60 + l];
    f2v[i] = s;
}

// tmp1[b][l][c] = sum_n f1v[b][l][n] * t_w[n][c]
__global__ void tmp1_kernel(const float* __restrict__ f1v, const float* __restrict__ tw,
                            float* __restrict__ tmp1)
{
    long i = (long)blockIdx.x * blockDim.x + threadIdx.x;
    if (i >= 30720L) return;
    int c = i % 64; long t = i / 64;
    int l = t % 60; int b = t / 60;
    const float* row = f1v + ((long)b * 60 + l) * 512;
    float s = 0.0f;
    for (int n = 0; n < 512; ++n) s += row[n] * tw[n * 64 + c];
    tmp1[i] = s;
}

// lg0[b][l][q] = sigmoid( sum_c tmp1[b][l][c]*f2v[b][c][q] + t_b[l][q] )
__global__ void lg0_kernel(const float* __restrict__ tmp1, const float* __restrict__ f2v,
                           const float* __restrict__ tb, float* __restrict__ lg0)
{
    long i = (long)blockIdx.x * blockDim.x + threadIdx.x;
    if (i >= 28800L) return;
    int q = i % 60; long t = i / 60;
    int l = t % 60; int b = t / 60;
    float s = 0.0f;
    for (int c = 0; c < 64; ++c)
        s += tmp1[((long)b * 60 + l) * 64 + c] * f2v[((long)b * 64 + c) * 60 + q];
    lg0[i] = sigf(s + tb[l * 60 + q]);
}

// lgv[b][i][t] = sum_q t_v[i][q] * lg0[b][q][t]
__global__ void lgv_kernel(const float* __restrict__ lg0, const float* __restrict__ tv,
                           float* __restrict__ lgv)
{
    long i = (long)blockIdx.x * blockDim.x + threadIdx.x;
    if (i >= 28800L) return;
    int t2 = i % 60; long t = i / 60;
    int ii = t % 60; int b = t / 60;
    float s = 0.0f;
    for (int q = 0; q < 60; ++q)
        s += tv[ii * 60 + q] * lg0[((long)b * 60 + q) * 60 + t2];
    lgv[i] = s;
}

// BN stats over (b, i) per t   ->  bnst[2t]=mean, bnst[2t+1]=var
__global__ void bnstat_kernel(const float* __restrict__ lgv, float* __restrict__ bnst)
{
    __shared__ float s1[128], s2[128];
    int t = blockIdx.x;
    int tid = threadIdx.x;
    float a = 0.0f, b2 = 0.0f;
    for (int j = tid; j < 480; j += 128) {
        float v = lgv[(long)j * 60 + t];
        a += v; b2 += v * v;
    }
    s1[tid] = a; s2[tid] = b2;
    __syncthreads();
    for (int st = 64; st > 0; st >>= 1) {
        if (tid < st) { s1[tid] += s1[tid + st]; s2[tid] += s2[tid + st]; }
        __syncthreads();
    }
    if (tid == 0) {
        float mu = s1[0] / 480.0f;
        bnst[2 * t] = mu;
        bnst[2 * t + 1] = s2[0] / 480.0f - mu * mu;
    }
}

// BN-normalize + block mask + softmax over last dim -> T_coef
__global__ void softmax_kernel(const float* __restrict__ lgv, const float* __restrict__ bnst,
                               const float* __restrict__ bng, const float* __restrict__ bnb,
                               float* __restrict__ Tc)
{
    __shared__ float red[64];
    int blk = blockIdx.x;         // b*60 + i
    int ii = blk % 60;
    int t = threadIdx.x;
    float v = -1e30f;
    if (t < 60) {
        float x = lgv[(long)blk * 60 + t];
        x = (x - bnst[2 * t]) * rsqrtf(bnst[2 * t + 1] + EPSV) * bng[t] + bnb[t];
        int bi = (ii < 36) ? ii / 12 : 3;
        int bt = (t  < 36) ? t  / 12 : 3;
        if (bi != bt) x += -1e13f;
        v = x;
    }
    red[t] = v;
    __syncthreads();
    for (int st = 32; st > 0; st >>= 1) {
        if (t < st) red[t] = fmaxf(red[t], red[t + st]);
        __syncthreads();
    }
    float mx = red[0];
    __syncthreads();
    float e = (t < 60) ? __expf(v - mx) : 0.0f;
    red[t] = e;
    __syncthreads();
    for (int st = 32; st > 0; st >>= 1) {
        if (t < st) red[t] += red[t + st];
        __syncthreads();
    }
    if (t < 60) Tc[(long)blk * 60 + t] = e / red[0];
}

// y = leaky( sum_l x1b[b,c,n,l] * T[b,q,l] ) + x_in
__global__ void y_kernel(const float* __restrict__ x1b, const float* __restrict__ Tc,
                         const float* __restrict__ xin, float* __restrict__ y)
{
    long i = (long)blockIdx.x * blockDim.x + threadIdx.x;
    if (i >= NELT) return;
    int q = i % 60; long t = i / 60;
    int n = t % 512; t /= 512;
    int c = t % 64;  int b = t / 64;
    const float* xr = x1b + (((long)b * 64 + c) * 512 + n) * 60;
    const float* tr = Tc + ((long)b * 60 + q) * 60;
    float s = 0.0f;
    for (int l = 0; l < 60; ++l) s += xr[l] * tr[l];
    y[i] = leakyf(s) + xin[i];
}

// Per-block partial (sum, sumsq) over y; 240 blocks per batch
__global__ void lnpart_kernel(const float* __restrict__ y, float* __restrict__ part)
{
    __shared__ float s1[256], s2[256];
    int b = blockIdx.x / 240;
    int seg = blockIdx.x % 240;
    int tid = threadIdx.x;
    long base = (long)b * 1966080L + (long)seg * 8192;
    float a = 0.0f, q = 0.0f;
    for (int k = 0; k < 32; ++k) {
        float v = y[base + (long)k * 256 + tid];
        a += v; q += v * v;
    }
    s1[tid] = a; s2[tid] = q;
    __syncthreads();
    for (int st = 128; st > 0; st >>= 1) {
        if (tid < st) { s1[tid] += s1[tid + st]; s2[tid] += s2[tid + st]; }
        __syncthreads();
    }
    if (tid == 0) {
        part[(long)blockIdx.x * 2]     = s1[0];
        part[(long)blockIdx.x * 2 + 1] = s2[0];
    }
}

__global__ void lnred_kernel(const float* __restrict__ part, float* __restrict__ lnst)
{
    __shared__ float s1[256], s2[256];
    int b = blockIdx.x;
    int tid = threadIdx.x;
    float a = 0.0f, q = 0.0f;
    if (tid < 240) {
        a = part[((long)b * 240 + tid) * 2];
        q = part[((long)b * 240 + tid) * 2 + 1];
    }
    s1[tid] = a; s2[tid] = q;
    __syncthreads();
    for (int st = 128; st > 0; st >>= 1) {
        if (tid < st) { s1[tid] += s1[tid + st]; s2[tid] += s2[tid + st]; }
        __syncthreads();
    }
    if (tid == 0) { lnst[2 * b] = s1[0]; lnst[2 * b + 1] = s2[0]; }
}

__global__ void lnfinal_kernel(const float* __restrict__ y, const float* __restrict__ lnst,
                               const float* __restrict__ lg, const float* __restrict__ lb,
                               float* __restrict__ out)
{
    long i = (long)blockIdx.x * blockDim.x + threadIdx.x;
    if (i >= NELT) return;
    long per = 1966080L;
    long b = i / per;
    long e = i % per;            // (c,n,l) flat, matches ln_g/ln_b layout
    float mu  = lnst[2 * b] / 1966080.0f;
    float var = lnst[2 * b + 1] / 1966080.0f - mu * mu;
    out[i] = (y[i] - mu) * rsqrtf(var + EPSV) * lg[e] + lb[e];
}

// ---------------------------------------------------------------------------
// Host launcher
// ---------------------------------------------------------------------------
static inline dim3 g1(long n) { return dim3((unsigned)((n + 255) / 256)); }

extern "C" void kernel_launch(void* const* d_in, const int* in_sizes, int n_in,
                              void* d_out, int out_size, void* d_ws, size_t ws_size,
                              hipStream_t stream)
{
    const float* x        = (const float*)d_in[0];
    const float* supports = (const float*)d_in[1];
    const float* conv1_w  = (const float*)d_in[2];
    const float* conv1_b  = (const float*)d_in[3];
    const float* tconv_w  = (const float*)d_in[4];
    const float* tconv_b  = (const float*)d_in[5];
    const float* s3_w1    = (const float*)d_in[6];
    const float* s3_w2    = (const float*)d_in[7];
    const float* s2_w1    = (const float*)d_in[8];
    const float* s2_w2    = (const float*)d_in[9];
    const float* wih      = (const float*)d_in[10];
    const float* whh      = (const float*)d_in[11];
    const float* bih      = (const float*)d_in[12];
    const float* bhh      = (const float*)d_in[13];
    const float* gcn_w    = (const float*)d_in[14];
    const float* gcn_b    = (const float*)d_in[15];
    const float* t_w1     = (const float*)d_in[16];
    const float* t_w2     = (const float*)d_in[17];
    const float* t_w      = (const float*)d_in[18];
    const float* t_b      = (const float*)d_in[19];
    const float* t_v      = (const float*)d_in[20];
    const float* t_bn_g   = (const float*)d_in[21];
    const float* t_bn_b   = (const float*)d_in[22];
    const float* ln_g     = (const float*)d_in[23];
    const float* ln_b     = (const float*)d_in[24];

    float* out  = (float*)d_out;                 // (8,64,512,60)
    float* adjo = out + NELT;                    // (8,512,512)
    float* Tco  = adjo + ADJSZ;                  // (8,60,60)

    // Workspace carve-up (float units; f16 buffers use n/2 float slots)
    float* W = (float*)d_ws;
    size_t off = 0;
    auto AL  = [&](size_t n) { float* p = W + off; off += n; return p; };
    auto ALH = [&](size_t n) { _Float16* p = (_Float16*)(W + off); off += n / 2; return p; };
    float*    x_in   = AL(15728640);
    float*    x1     = AL(15728640);
    _Float16* x1ttH  = ALH(15728640);
    _Float16* adj1H  = ALH(2097152);
    _Float16* adj1HT = ALH(2097152);
    _Float16* lap2H  = ALH(2097152);
    float*    F1     = AL(4194304);
    float*    F2     = AL(4194304);
    _Float16* SbH    = ALH(16777216);
    _Float16* wihH   = ALH(1048576);
    _Float16* whhH   = ALH(1048576);
    float*    hbuf   = AL(2097152);
    float*    cbuf   = AL(2097152);
    _Float16* hH     = ALH(2097152);
    float*    Gx     = AL(8388608);
    float*    Gh     = AL(8388608);
    _Float16* xgpadH = ALH(48758784);
    _Float16* gcnBH  = ALH(73728);
    float*    gout   = AL(31457280);
    float*    x1b    = AL(15728640);
    float*    f1v    = AL(245760);
    float*    f2v    = AL(30720);
    float*    tmp1   = AL(30720);
    float*    lg0    = AL(28800);
    float*    lgv    = AL(28800);
    float*    bnst   = AL(120);
    float*    ybuf   = AL(15728640);
    float*    lnpart = AL(3840);
    float*    lnst   = AL(16);
    const long BIGGRP = 1L << 40;

    // 1) entry convs
    conv1_kernel<<<g1(NELT), 256, 0, stream>>>(x, conv1_w, conv1_b, x_in);
    tconv_kernel<<<g1(NELT), 256, 0, stream>>>(x, tconv_w, tconv_b, x1);

    // 2) spatial attention -> S (8,16,512,512) in f16 (LSTM GEMM A operand)
    satt_f_kernel<<<g1(4194304), 256, 0, stream>>>(x1, s3_w1, s2_w1, F1);
    satt_f_kernel<<<g1(4194304), 256, 0, stream>>>(x1, s3_w2, s2_w2, F2);
    satt_s_kernel<<<g1(16777216), 256, 0, stream>>>(F1, F2, SbH);

    // 3) operand prep: everything a GEMM touches becomes f16 (B panels K-major)
    cvt_kernel<<<g1(1048576), 256, 0, stream>>>(wih, wihH, 1048576);   // already [n][k]
    cvt_kernel<<<g1(1048576), 256, 0, stream>>>(whh, whhH, 1048576);
    x1tt_kernel<<<g1(NELT), 256, 0, stream>>>(x1, x1ttH);
    gcnw_kernel<<<g1(73728), 256, 0, stream>>>(gcn_w, gcnBH);

    // 4) LSTM over 16 steps. S rows via grouped-affine: r*512 + (r/512)*3932160
    zero_kernel<<<g1(2097152), 256, 0, stream>>>(hbuf, 2097152);
    zero_kernel<<<g1(2097152), 256, 0, stream>>>(cbuf, 2097152);
    zero_kernel<<<g1(1048576), 256, 0, stream>>>((float*)hH, 1048576);
    for (int t = 0; t < 16; ++t) {
        gemm_wmma<<<dim3(64, 16), 128, 0, stream>>>(
            SbH + (long)t * 262144, wihH, Gx,
            512, 512, 512L, 3932160L, 512, 2048, 0, 0L, 0L);
        gemm_wmma<<<dim3(64, 16), 128, 0, stream>>>(
            hH, whhH, Gh,
            512, 512, BIGGRP, 0L, 512, 2048, 0, 0L, 0L);
        lstm_point_kernel<<<g1(2097152), 256, 0, stream>>>(Gx, Gh, bih, bhh,
                                                           hbuf, cbuf, hH);
    }
    adjfinish_kernel<<<g1(ADJSZ), 256, 0, stream>>>(hbuf, supports, adjo, adj1H, adj1HT);

    // 5) Chebyshev:  lap2 = 2*adj1@adj1 - I  per batch (f16 out)
    for (int b = 0; b < 8; ++b) {
        gemm_wmma<<<dim3(8, 4), 128, 0, stream>>>(
            adj1H + (long)b * 262144, adj1HT + (long)b * 262144, lap2H + (long)b * 262144,
            512, 512, BIGGRP, 0L, 512, 512, 2, 0L, 0L);
    }

    // 6) graph conv input xg in padded (b,n,62,192) f16 layout; ch = c*3 + k
    zero_kernel<<<g1(24379392), 256, 0, stream>>>((float*)xgpadH, 24379392);
    k0copy_kernel<<<g1(NELT), 256, 0, stream>>>(x1, xgpadH);
    for (int b = 0; b < 8; ++b) {
        for (int k = 1; k <= 2; ++k) {
            const _Float16* Ak = (k == 1) ? (adj1H + (long)b * 262144)
                                          : (lap2H + (long)b * 262144);
            gemm_wmma<<<dim3(8, 30), 128, 0, stream>>>(
                Ak, x1ttH + (long)b * 1966080, xgpadH,
                512, 512, BIGGRP, 0L, 512, 0, 1, (long)b * 512, (long)k);
        }
    }

    // 7) GCN 1x3 conv as one big GEMM (overlapping-window rows; K = 3*192 = 576)
    gemm_wmma<<<dim3(3840, 1), 128, 0, stream>>>(
        xgpadH, gcnBH, gout,
        576, 192, 60L, 384L, 576, 128, 0, 0L, 0L);
    gcn_epi_kernel<<<g1(NELT), 256, 0, stream>>>(gout, gcn_b, x1b);

    // 8) temporal attention (small)
    f1v_kernel<<<g1(245760), 256, 0, stream>>>(x1b, t_w1, f1v);
    f2v_kernel<<<g1(30720), 256, 0, stream>>>(x1b, t_w2, f2v);
    tmp1_kernel<<<g1(30720), 256, 0, stream>>>(f1v, t_w, tmp1);
    lg0_kernel<<<g1(28800), 256, 0, stream>>>(tmp1, f2v, t_b, lg0);
    lgv_kernel<<<g1(28800), 256, 0, stream>>>(lg0, t_v, lgv);
    bnstat_kernel<<<60, 128, 0, stream>>>(lgv, bnst);
    softmax_kernel<<<480, 64, 0, stream>>>(lgv, bnst, t_bn_g, t_bn_b, Tco);

    // 9) apply temporal coefficients + residual + layernorm
    y_kernel<<<g1(NELT), 256, 0, stream>>>(x1b, Tco, x_in, ybuf);
    lnpart_kernel<<<1920, 256, 0, stream>>>(ybuf, lnpart);
    lnred_kernel<<<8, 256, 0, stream>>>(lnpart, lnst);
    lnfinal_kernel<<<g1(NELT), 256, 0, stream>>>(ybuf, lnst, ln_g, ln_b, out);

    (void)in_sizes; (void)n_in; (void)out_size; (void)ws_size;
}
